// MultiHeadAttention_79723182949059
// MI455X (gfx1250) — matrix operand order repaired and checked by reading, not compiled, not run on hardware
//
#include <hip/hip_runtime.h>
#include <hip/hip_bf16.h>
#include <hip/hip_fp16.h>

// MI455X / gfx1250, wave32. Fused MHA forward:
//   3x projection GEMM (f16 WMMA, f32 accum) -> flash attention (online softmax,
//   causal mask analytic) -> output projection GEMM (f32 out).
// All matrix math through v_wmma_f32_16x16x32_f16.

typedef __attribute__((ext_vector_type(16))) _Float16 v16h;
typedef __attribute__((ext_vector_type(8)))  _Float16 v8h;
typedef __attribute__((ext_vector_type(8)))  float    v8f;

#define B_  2
#define L_  2048
#define D_  1024
#define H_  16
#define DH_ 64

static __device__ __forceinline__ v8f wmma16x16x32(v16h a, v16h b, v8f c) {
  // D = A(16x32 f16) * B(32x16 f16) + C(16x16 f32)
  return __builtin_amdgcn_wmma_f32_16x16x32_f16(
      /*neg_a=*/false, a, /*neg_b=*/false, b,
      /*c_mod=*/(short)0, c, /*reuse_a=*/false, /*reuse_b=*/false);
}

// A-matrix fragment (16x32, f16) from row-major [m, k] data (lda in elements).
// ISA layout: lanes 0-15 hold M=0..15 with K groups {0..7,16..23};
//             lanes 16-31 hold the same M with K groups {8..15,24..31}.
static __device__ __forceinline__ v16h load_a_frag(const _Float16* base, int lda, int lane) {
  const int m    = lane & 15;
  const int kmid = (lane >> 4) << 3;   // 0 or 8
  const _Float16* p = base + (size_t)m * lda + kmid;
  union { v16h v; v8h h[2]; } u;
  u.h[0] = *(const v8h*)(p);           // K = kmid + 0..7
  u.h[1] = *(const v8h*)(p + 16);      // K = 16 + kmid + 0..7
  return u.v;
}

// B-matrix fragment (32x16, f16) where memory is laid out n-major: row n holds
// contiguous k. Layout: lanes 0-15 hold K groups {0..7,16..23}, lanes 16-31 {8..15,24..31}.
static __device__ __forceinline__ v16h load_b_frag(const _Float16* rowbase, int ldb, int lane) {
  const int n  = lane & 15;
  const int kh = (lane >> 4) << 3;     // 0 or 8
  const _Float16* p = rowbase + (size_t)n * ldb + kh;
  union { v16h v; v8h h[2]; } u;
  u.h[0] = *(const v8h*)(p);
  u.h[1] = *(const v8h*)(p + 16);
  return u.v;
}

// ---------------------------------------------------------------- conversions
__global__ __launch_bounds__(256) void cvt_f32_f16_kernel(
    const float* __restrict__ src, _Float16* __restrict__ dst, int n) {
  int i = blockIdx.x * blockDim.x + threadIdx.x;
  if (i < n) dst[i] = (_Float16)src[i];
}

// Wt[n][k] = (f16) W[k][n]; W is [dim, dim] row-major fp32.
__global__ __launch_bounds__(256) void transpose_cvt_kernel(
    const float* __restrict__ W, _Float16* __restrict__ Wt, int dim) {
  int i = blockIdx.x * blockDim.x + threadIdx.x;
  if (i < dim * dim) {
    int k = i / dim, n = i - k * dim;
    Wt[(size_t)n * dim + k] = (_Float16)W[i];
  }
}

// ---------------------------------------------------------------- projections
// C[m,n] = sum_k A[m,k]*Bt[n,k] + bias[n].
// mode 0: f16 out, [B,H,L,DH] layout (Q, K)
// mode 1: f16 out, [B,H,DH,L] layout (V transposed, feeds P*V B-frags)
// mode 2: f32 out, row-major [M,N] (final projection -> d_out)
__global__ __launch_bounds__(256) void wmma_gemm_kernel(
    const _Float16* __restrict__ A, const _Float16* __restrict__ Bt,
    const float* __restrict__ bias,
    _Float16* __restrict__ outh, float* __restrict__ outf,
    int M, int Kd, int Nd, int mode)
{
  const int lane = threadIdx.x & 31;
  const int wave = threadIdx.x >> 5;
  const int m0 = blockIdx.y * 16;
  const int n0 = blockIdx.x * 256 + wave * 32;   // each wave: 16x32 output
  if (m0 >= M || n0 >= Nd) return;

  v8f c0 = {};
  v8f c1 = {};
  const _Float16* arow  = A  + (size_t)m0 * Kd;
  const _Float16* b0row = Bt + (size_t)n0 * Kd;
  const _Float16* b1row = Bt + (size_t)(n0 + 16) * Kd;
  for (int k = 0; k < Kd; k += 32) {
    v16h a  = load_a_frag(arow + k,  Kd, lane);
    v16h b0 = load_b_frag(b0row + k, Kd, lane);
    v16h b1 = load_b_frag(b1row + k, Kd, lane);
    c0 = wmma16x16x32(a, b0, c0);
    c1 = wmma16x16x32(a, b1, c1);
  }

  const int ncol  = lane & 15;
  const int mbase = (lane >> 4) << 3;            // C/D layout: m = mbase + r
  const float bias0 = bias[n0 + ncol];
  const float bias1 = bias[n0 + 16 + ncol];
  const int ng0 = n0 + ncol, ng1 = n0 + 16 + ncol;
#pragma unroll
  for (int r = 0; r < 8; ++r) {
    const int mg = m0 + mbase + r;
    const float v0 = c0[r] + bias0;
    const float v1 = c1[r] + bias1;
    if (mode == 2) {
      outf[(size_t)mg * Nd + ng0] = v0;
      outf[(size_t)mg * Nd + ng1] = v1;
    } else {
      const int b = mg >> 11, l = mg & (L_ - 1);  // M = B*L rows
      if (mode == 0) {
        outh[(((size_t)b * H_ + (ng0 >> 6)) * L_ + l) * DH_ + (ng0 & 63)] = (_Float16)v0;
        outh[(((size_t)b * H_ + (ng1 >> 6)) * L_ + l) * DH_ + (ng1 & 63)] = (_Float16)v1;
      } else {
        outh[(((size_t)b * H_ + (ng0 >> 6)) * DH_ + (ng0 & 63)) * L_ + l] = (_Float16)v0;
        outh[(((size_t)b * H_ + (ng1 >> 6)) * DH_ + (ng1 & 63)) * L_ + l] = (_Float16)v1;
      }
    }
  }
}

// ---------------------------------------------------------------- attention
// One wave per 16-query tile. Streams causal K/V in 32-key steps, online
// softmax in f32, S and P*V via WMMA. P re-striped D-layout -> A-layout
// through a private per-wave LDS tile (same-wave DS ordering, no barrier).
__global__ __launch_bounds__(128) void flash_attn_kernel(
    const _Float16* __restrict__ Q,   // [B,H,L,DH]
    const _Float16* __restrict__ K,   // [B,H,L,DH]
    const _Float16* __restrict__ Vt,  // [B,H,DH,L]
    _Float16* __restrict__ O)         // [B,L,H*DH]
{
  __shared__ __align__(64) _Float16 plds[4][16 * 32];
  const int lane = threadIdx.x & 31;
  const int wave = threadIdx.x >> 5;
  const int tile = blockIdx.x * 4 + wave;        // 0 .. B*H*(L/16)-1
  const int qt = tile & (L_ / 16 - 1);
  const int bh = tile >> 7;                      // L_/16 == 128
  const int q0 = qt << 4;

  const _Float16* Qb = Q  + (size_t)bh * L_ * DH_;
  const _Float16* Kb = K  + (size_t)bh * L_ * DH_;
  const _Float16* Vb = Vt + (size_t)bh * DH_ * L_;

  const int ncol  = lane & 15;
  const int mbase = (lane >> 4) << 3;

  // Q fragment is loop-invariant: 16 rows x 64 d -> two A-frags (k-steps of 32)
  const v16h aq0 = load_a_frag(Qb + (size_t)q0 * DH_ + 0,  DH_, lane);
  const v16h aq1 = load_a_frag(Qb + (size_t)q0 * DH_ + 32, DH_, lane);

  v8f acc0 = {}, acc1 = {}, acc2 = {}, acc3 = {};   // O tile: 16 x 64
  float mrow[8], lrow[8];
#pragma unroll
  for (int r = 0; r < 8; ++r) { mrow[r] = -1e30f; lrow[r] = 0.0f; }

  _Float16* pw = plds[wave];
  const int kv_end = q0 + 16;                     // causal: kv0 <= q0+15
  for (int kv0 = 0; kv0 < kv_end; kv0 += 32) {
    // ---- S = Q * K^T over d = 64 (two k-steps), two 16-key column tiles
    v8f s0 = {}, s1 = {};
    {
      v16h bk00 = load_b_frag(Kb + (size_t)(kv0)      * DH_ + 0,  DH_, lane);
      v16h bk10 = load_b_frag(Kb + (size_t)(kv0 + 16) * DH_ + 0,  DH_, lane);
      s0 = wmma16x16x32(aq0, bk00, s0);
      s1 = wmma16x16x32(aq0, bk10, s1);
      v16h bk01 = load_b_frag(Kb + (size_t)(kv0)      * DH_ + 32, DH_, lane);
      v16h bk11 = load_b_frag(Kb + (size_t)(kv0 + 16) * DH_ + 32, DH_, lane);
      s0 = wmma16x16x32(aq1, bk01, s0);
      s1 = wmma16x16x32(aq1, bk11, s1);
    }
    if (kv0 + 32 < kv_end) {                      // global_prefetch_b8 next K tile
      __builtin_prefetch(Kb + (size_t)(kv0 + 32) * DH_, 0, 1);
    }

    // ---- scale + causal mask + online softmax (row stats across 16 lanes)
#pragma unroll
    for (int r = 0; r < 8; ++r) {
      const int mg = q0 + mbase + r;
      float v0 = s0[r] * 0.125f;                  // 1/sqrt(DH)
      float v1 = s1[r] * 0.125f;
      if (kv0 + ncol      > mg) v0 = -1e12f;      // matches reference -NEG_INF
      if (kv0 + 16 + ncol > mg) v1 = -1e12f;
      float mx = fmaxf(v0, v1);
      mx = fmaxf(mx, __shfl_xor(mx, 8, 32));
      mx = fmaxf(mx, __shfl_xor(mx, 4, 32));
      mx = fmaxf(mx, __shfl_xor(mx, 2, 32));
      mx = fmaxf(mx, __shfl_xor(mx, 1, 32));
      const float mnew  = fmaxf(mrow[r], mx);
      const float alpha = __expf(mrow[r] - mnew);
      const float p0 = __expf(v0 - mnew);
      const float p1 = __expf(v1 - mnew);
      float rs = p0 + p1;
      rs += __shfl_xor(rs, 8, 32);
      rs += __shfl_xor(rs, 4, 32);
      rs += __shfl_xor(rs, 2, 32);
      rs += __shfl_xor(rs, 1, 32);
      lrow[r] = lrow[r] * alpha + rs;
      mrow[r] = mnew;
      acc0[r] *= alpha; acc1[r] *= alpha; acc2[r] *= alpha; acc3[r] *= alpha;
      // stage P (D-layout element (m, n)) into row-major LDS tile [16][32]
      pw[(mbase + r) * 32 + ncol]      = (_Float16)p0;
      pw[(mbase + r) * 32 + 16 + ncol] = (_Float16)p1;
    }
    asm volatile("" ::: "memory");                // keep ds_load after ds_store

    // ---- O += P * V  (A-frag from LDS, B-frags from transposed V)
    v16h ap  = load_a_frag(pw, 32, lane);
    v16h bv0 = load_b_frag(Vb + (size_t)( 0) * L_ + kv0, L_, lane);
    v16h bv1 = load_b_frag(Vb + (size_t)(16) * L_ + kv0, L_, lane);
    v16h bv2 = load_b_frag(Vb + (size_t)(32) * L_ + kv0, L_, lane);
    v16h bv3 = load_b_frag(Vb + (size_t)(48) * L_ + kv0, L_, lane);
    acc0 = wmma16x16x32(ap, bv0, acc0);
    acc1 = wmma16x16x32(ap, bv1, acc1);
    acc2 = wmma16x16x32(ap, bv2, acc2);
    acc3 = wmma16x16x32(ap, bv3, acc3);
    asm volatile("" ::: "memory");                // wmma consumed before rewrite
  }

  // ---- normalize and store as [B, L, H*DH] f16 (feeds final projection)
  const int b = bh >> 4, h = bh & 15;
#pragma unroll
  for (int r = 0; r < 8; ++r) {
    const int mg = q0 + mbase + r;
    const float rinv = 1.0f / lrow[r];
    _Float16* orow = O + ((size_t)(b * L_ + mg)) * D_ + h * DH_ + ncol;
    orow[0]  = (_Float16)(acc0[r] * rinv);
    orow[16] = (_Float16)(acc1[r] * rinv);
    orow[32] = (_Float16)(acc2[r] * rinv);
    orow[48] = (_Float16)(acc3[r] * rinv);
  }
}

// ---------------------------------------------------------------- launch
extern "C" void kernel_launch(void* const* d_in, const int* in_sizes, int n_in,
                              void* d_out, int out_size, void* d_ws, size_t ws_size,
                              hipStream_t stream) {
  const float* q  = (const float*)d_in[0];
  const float* k  = (const float*)d_in[1];
  const float* v  = (const float*)d_in[2];
  // d_in[3] = causal mask: recomputed analytically in-kernel
  const float* Wq = (const float*)d_in[4];
  const float* bq = (const float*)d_in[5];
  const float* Wk = (const float*)d_in[6];
  const float* bk = (const float*)d_in[7];
  const float* Wv = (const float*)d_in[8];
  const float* bv = (const float*)d_in[9];
  const float* Wo = (const float*)d_in[10];
  const float* bo = (const float*)d_in[11];
  float* out = (float*)d_out;

  const size_t MT = (size_t)B_ * L_;     // 4096 rows
  const size_t XE = MT * D_;             // 4 Mi elems per activation tensor
  const size_t WE = (size_t)D_ * D_;     // 1 Mi elems per weight

  _Float16* ws  = (_Float16*)d_ws;       // 56 MB total
  _Float16* Xq  = ws;                    // reused as attention output later
  _Float16* Xk  = ws + XE;
  _Float16* Xv  = ws + 2 * XE;
  _Float16* WqT = ws + 3 * XE;
  _Float16* WkT = WqT + WE;
  _Float16* WvT = WkT + WE;
  _Float16* WoT = WvT + WE;
  _Float16* Qh  = WoT + WE;              // [B,H,L,DH]
  _Float16* Kh  = Qh + XE;               // [B,H,L,DH]
  _Float16* Vth = Kh + XE;               // [B,H,DH,L]
  _Float16* AO  = Xq;                    // [B,L,D] attention output (reuses Xq)

  const dim3 cb(256);
  cvt_f32_f16_kernel<<<dim3((unsigned)((XE + 255) / 256)), cb, 0, stream>>>(q, Xq, (int)XE);
  cvt_f32_f16_kernel<<<dim3((unsigned)((XE + 255) / 256)), cb, 0, stream>>>(k, Xk, (int)XE);
  cvt_f32_f16_kernel<<<dim3((unsigned)((XE + 255) / 256)), cb, 0, stream>>>(v, Xv, (int)XE);
  transpose_cvt_kernel<<<dim3((unsigned)((WE + 255) / 256)), cb, 0, stream>>>(Wq, WqT, D_);
  transpose_cvt_kernel<<<dim3((unsigned)((WE + 255) / 256)), cb, 0, stream>>>(Wk, WkT, D_);
  transpose_cvt_kernel<<<dim3((unsigned)((WE + 255) / 256)), cb, 0, stream>>>(Wv, WvT, D_);
  transpose_cvt_kernel<<<dim3((unsigned)((WE + 255) / 256)), cb, 0, stream>>>(Wo, WoT, D_);

  const dim3 gg(D_ / 256, (unsigned)(MT / 16));   // (4, 256), 8 waves/block
  wmma_gemm_kernel<<<gg, dim3(256), 0, stream>>>(Xq, WqT, bq, Qh,  nullptr, (int)MT, D_, D_, 0);
  wmma_gemm_kernel<<<gg, dim3(256), 0, stream>>>(Xk, WkT, bk, Kh,  nullptr, (int)MT, D_, D_, 0);
  wmma_gemm_kernel<<<gg, dim3(256), 0, stream>>>(Xv, WvT, bv, Vth, nullptr, (int)MT, D_, D_, 1);

  flash_attn_kernel<<<dim3((B_ * H_ * (L_ / 16)) / 4), dim3(128), 0, stream>>>(Qh, Kh, Vth, AO);

  wmma_gemm_kernel<<<gg, dim3(256), 0, stream>>>(AO, WoT, bo, nullptr, out, (int)MT, D_, D_, 2);
}
